// Attention_7894149890571
// MI455X (gfx1250) — matrix operand order repaired and checked by reading, not compiled, not run on hardware
//
#include <hip/hip_runtime.h>

typedef __bf16 bf16;
typedef __attribute__((ext_vector_type(16))) __bf16 v16bf;
typedef __attribute__((ext_vector_type(8)))  __bf16 v8bf;
typedef __attribute__((ext_vector_type(8)))  float  v8f;

#define CDIM   1024
#define LSEQ   2048
#define NHEADS 16
#define HD     64

__device__ __forceinline__ v8f wmma_bf16(v16bf a, v16bf b, v8f c) {
  return __builtin_amdgcn_wmma_f32_16x16x32_bf16(false, a, false, b, (short)0, c, false, false);
}

// A fragment (16x32, row-major source, elem stride in bf16):
//   elems e=0..15 : A[m][16*lh + e]
__device__ __forceinline__ v16bf load_frag_pair(const bf16* p0, const bf16* p1) {
  union { v16bf v; v8bf h[2]; } u;
  u.h[0] = *(const v8bf*)p0;
  u.h[1] = *(const v8bf*)p1;
  return u.v;
}
__device__ __forceinline__ v16bf loadA(const bf16* base, int m, int lh, int stride) {
  return load_frag_pair(base + m * stride + 16 * lh,
                        base + m * stride + 16 * lh + 8);
}
// B fragment from N-major storage Bt[n][k] (stride in bf16):
//   elems e=0..15 : Bt[n][16*lh + e]
__device__ __forceinline__ v16bf loadB(const bf16* base, int n, int lh, int stride) {
  return load_frag_pair(base + n * stride + 16 * lh,
                        base + n * stride + 16 * lh + 8);
}

// ---------------- Kernel 1: QKV GEMM + scatter to Q/K/Vt (bf16) --------------
// out[M=4096, N=3072] = x[M,1024] * qkv_w[N,1024]^T
__global__ __launch_bounds__(256)
void qkv_gemm_kernel(const float* __restrict__ x, const float* __restrict__ qkv_w,
                     const float* __restrict__ s_ptr,
                     bf16* __restrict__ Q, bf16* __restrict__ Kv, bf16* __restrict__ Vt)
{
  __shared__ bf16 Abuf[128 * 32];
  __shared__ bf16 Bbuf[64 * 32];
  const int tid  = threadIdx.x;
  const int lane = tid & 31, wid = tid >> 5;
  const int lh   = lane >> 4, idx = lane & 15;
  const int wm   = wid & 3,  wn  = wid >> 2;            // 4 (M) x 2 (N) waves
  const int rowBase = blockIdx.y * 128;
  const int colBase = blockIdx.x * 64;

  v8f acc[2][2] = {};
  for (int k0 = 0; k0 < CDIM; k0 += 32) {
    __syncthreads();
    for (int c = tid; c < 1024; c += 256) {             // 128x32 f32 -> bf16
      int r = c >> 3, c4 = (c & 7) << 2;
      const float4 f = *(const float4*)(x + (size_t)(rowBase + r) * CDIM + k0 + c4);
      bf16* d = Abuf + r * 32 + c4;
      d[0] = (bf16)f.x; d[1] = (bf16)f.y; d[2] = (bf16)f.z; d[3] = (bf16)f.w;
    }
    for (int c = tid; c < 512; c += 256) {              // 64x32 f32 -> bf16
      int r = c >> 3, c4 = (c & 7) << 2;
      const float4 f = *(const float4*)(qkv_w + (size_t)(colBase + r) * CDIM + k0 + c4);
      bf16* d = Bbuf + r * 32 + c4;
      d[0] = (bf16)f.x; d[1] = (bf16)f.y; d[2] = (bf16)f.z; d[3] = (bf16)f.w;
    }
    __syncthreads();
    v16bf a0 = loadA(Abuf + (wm * 32 +  0) * 32, idx, lh, 32);
    v16bf a1 = loadA(Abuf + (wm * 32 + 16) * 32, idx, lh, 32);
    v16bf b0 = loadB(Bbuf + (wn * 32 +  0) * 32, idx, lh, 32);
    v16bf b1 = loadB(Bbuf + (wn * 32 + 16) * 32, idx, lh, 32);
    acc[0][0] = wmma_bf16(a0, b0, acc[0][0]);
    acc[0][1] = wmma_bf16(a0, b1, acc[0][1]);
    acc[1][0] = wmma_bf16(a1, b0, acc[1][0]);
    acc[1][1] = wmma_bf16(a1, b1, acc[1][1]);
  }
  // combined scale for Q: qk_scale * s * ln(L) = 0.125 * s * ln(2048)
  const float cs = 0.125f * s_ptr[0] * 7.6246189861593985f;
  #pragma unroll
  for (int i = 0; i < 2; i++)
    #pragma unroll
    for (int j = 0; j < 2; j++)
      #pragma unroll
      for (int r = 0; r < 8; r++) {
        int gr = rowBase + wm * 32 + i * 16 + r + 8 * lh;
        int gc = colBase + wn * 32 + j * 16 + idx;
        float v = acc[i][j][r];
        int b = gr >> 11, l = gr & 2047;
        int sel = gc >> 10, c = gc & 1023;        // uniform across the tile
        int h = c >> 6, d = c & 63;
        size_t bh = (size_t)(b * NHEADS + h);
        if (sel == 0)      Q [(bh * LSEQ + l) * HD + d]  = (bf16)(v * cs);
        else if (sel == 1) Kv[(bh * LSEQ + l) * HD + d]  = (bf16)v;
        else               Vt[(bh * HD + d) * LSEQ + l]  = (bf16)v;
      }
}

// ---------------- Kernel 2: flash attention, one wave per 16-query tile ------
__global__ __launch_bounds__(128)
void attn_kernel(const bf16* __restrict__ Q, const bf16* __restrict__ Kv,
                 const bf16* __restrict__ Vt, const float* __restrict__ mask,
                 bf16* __restrict__ Oo)
{
  __shared__ bf16 pshared[4][16 * 32];
  const int tid  = threadIdx.x;
  const int lane = tid & 31, wid = tid >> 5;
  const int lh   = lane >> 4, idx = lane & 15;
  const int gw   = blockIdx.x * 4 + wid;                // 0..4095
  const int b    = gw >> 11;
  const int head = (gw >> 7) & 15;
  const int q0   = (gw & 127) << 4;
  const size_t bh = (size_t)(b * NHEADS + head);
  const bf16*  Qb = Q  + (bh * LSEQ + q0) * HD;
  const bf16*  Kb = Kv + bh * LSEQ * HD;
  const bf16*  Vb = Vt + bh * HD * LSEQ;
  const float* Mb = mask + (bh * LSEQ + q0) * LSEQ;
  bf16* pbuf = pshared[wid];

  v16bf qf[2];
  #pragma unroll
  for (int kk = 0; kk < 2; kk++)
    qf[kk] = load_frag_pair(Qb + idx * HD + kk * 32 + 16 * lh,
                            Qb + idx * HD + kk * 32 + 16 * lh + 8);

  float rowM[8], rowL[8];
  v8f o[4] = {};
  #pragma unroll
  for (int r = 0; r < 8; r++) { rowM[r] = -1e30f; rowL[r] = 0.0f; }

  for (int kb = 0; kb < LSEQ / 32; kb++) {
    const int kc = kb * 32;
    // prefetch next key-block's mask rows (streaming data, hide HBM latency)
    if (kb + 1 < LSEQ / 32) {
      #pragma unroll
      for (int r = 0; r < 8; r++)
        __builtin_prefetch(Mb + (size_t)(r + 8 * lh) * LSEQ + kc + 32 + idx, 0, 0);
    }
    v8f s0 = {}, s1 = {};
    #pragma unroll
    for (int kk = 0; kk < 2; kk++) {
      v16bf bk0 = load_frag_pair(Kb + (size_t)(kc + idx) * HD + kk * 32 + 16 * lh,
                                 Kb + (size_t)(kc + idx) * HD + kk * 32 + 16 * lh + 8);
      v16bf bk1 = load_frag_pair(Kb + (size_t)(kc + 16 + idx) * HD + kk * 32 + 16 * lh,
                                 Kb + (size_t)(kc + 16 + idx) * HD + kk * 32 + 16 * lh + 8);
      s0 = wmma_bf16(qf[kk], bk0, s0);
      s1 = wmma_bf16(qf[kk], bk1, s1);
    }
    // add mask (non-temporal stream: 512MB > L2, keep K/Vt resident),
    // then online softmax (rows r+8*lh live in this lane's half)
    #pragma unroll
    for (int r = 0; r < 8; r++) {
      const float* mrow = Mb + (size_t)(r + 8 * lh) * LSEQ + kc;
      s0[r] += __builtin_nontemporal_load(mrow + idx);
      s1[r] += __builtin_nontemporal_load(mrow + 16 + idx);
      float bm = fmaxf(s0[r], s1[r]);
      bm = fmaxf(bm, __shfl_xor(bm, 1, 32));
      bm = fmaxf(bm, __shfl_xor(bm, 2, 32));
      bm = fmaxf(bm, __shfl_xor(bm, 4, 32));
      bm = fmaxf(bm, __shfl_xor(bm, 8, 32));
      float newM = fmaxf(rowM[r], bm);
      float corr = __expf(rowM[r] - newM);
      rowM[r] = newM;
      float e0 = __expf(s0[r] - newM);
      float e1 = __expf(s1[r] - newM);
      float bs = e0 + e1;
      bs += __shfl_xor(bs, 1, 32);
      bs += __shfl_xor(bs, 2, 32);
      bs += __shfl_xor(bs, 4, 32);
      bs += __shfl_xor(bs, 8, 32);
      rowL[r] = rowL[r] * corr + bs;
      o[0][r] *= corr; o[1][r] *= corr; o[2][r] *= corr; o[3][r] *= corr;
      pbuf[(r + 8 * lh) * 32 + idx]      = (bf16)e0;   // P tile, bf16
      pbuf[(r + 8 * lh) * 32 + 16 + idx] = (bf16)e1;
    }
    // reload P as A-fragment (same-wave LDS ops are in-order)
    v16bf pa = load_frag_pair(pbuf + idx * 32 + 16 * lh,
                              pbuf + idx * 32 + 16 * lh + 8);
    #pragma unroll
    for (int t = 0; t < 4; t++) {
      v16bf vb = load_frag_pair(Vb + (size_t)(16 * t + idx) * LSEQ + kc + 16 * lh,
                                Vb + (size_t)(16 * t + idx) * LSEQ + kc + 16 * lh + 8);
      o[t] = wmma_bf16(pa, vb, o[t]);
    }
  }
  #pragma unroll
  for (int r = 0; r < 8; r++) {
    float inv = 1.0f / rowL[r];
    int row = q0 + r + 8 * lh;
    bf16* orow = Oo + ((size_t)b * LSEQ + row) * CDIM + head * HD;
    #pragma unroll
    for (int t = 0; t < 4; t++)
      orow[16 * t + idx] = (bf16)(o[t][r] * inv);
  }
}

// ---------------- Kernel 3: output projection + bias -------------------------
// out[4096,1024] = Ain[4096,1024] * proj_w[1024,1024]^T + proj_b
__global__ __launch_bounds__(256)
void proj_gemm_kernel(const bf16* __restrict__ Ain, const float* __restrict__ W,
                      const float* __restrict__ bias, float* __restrict__ out)
{
  __shared__ bf16 Abuf[128 * 32];
  __shared__ bf16 Bbuf[64 * 32];
  const int tid  = threadIdx.x;
  const int lane = tid & 31, wid = tid >> 5;
  const int lh   = lane >> 4, idx = lane & 15;
  const int wm   = wid & 3,  wn  = wid >> 2;
  const int rowBase = blockIdx.y * 128;
  const int colBase = blockIdx.x * 64;

  v8f acc[2][2] = {};
  for (int k0 = 0; k0 < CDIM; k0 += 32) {
    __syncthreads();
    for (int c = tid; c < 512; c += 256) {              // 128x32 bf16 copy
      int r = c >> 2, c8 = (c & 3) << 3;
      *(v8bf*)(Abuf + r * 32 + c8) =
          *(const v8bf*)(Ain + (size_t)(rowBase + r) * CDIM + k0 + c8);
    }
    for (int c = tid; c < 512; c += 256) {              // 64x32 f32 -> bf16
      int r = c >> 3, c4 = (c & 7) << 2;
      const float4 f = *(const float4*)(W + (size_t)(colBase + r) * CDIM + k0 + c4);
      bf16* d = Bbuf + r * 32 + c4;
      d[0] = (bf16)f.x; d[1] = (bf16)f.y; d[2] = (bf16)f.z; d[3] = (bf16)f.w;
    }
    __syncthreads();
    v16bf a0 = loadA(Abuf + (wm * 32 +  0) * 32, idx, lh, 32);
    v16bf a1 = loadA(Abuf + (wm * 32 + 16) * 32, idx, lh, 32);
    v16bf b0 = loadB(Bbuf + (wn * 32 +  0) * 32, idx, lh, 32);
    v16bf b1 = loadB(Bbuf + (wn * 32 + 16) * 32, idx, lh, 32);
    acc[0][0] = wmma_bf16(a0, b0, acc[0][0]);
    acc[0][1] = wmma_bf16(a0, b1, acc[0][1]);
    acc[1][0] = wmma_bf16(a1, b0, acc[1][0]);
    acc[1][1] = wmma_bf16(a1, b1, acc[1][1]);
  }
  #pragma unroll
  for (int i = 0; i < 2; i++)
    #pragma unroll
    for (int j = 0; j < 2; j++)
      #pragma unroll
      for (int r = 0; r < 8; r++) {
        int gr = rowBase + wm * 32 + i * 16 + r + 8 * lh;
        int gc = colBase + wn * 32 + j * 16 + idx;
        // final output: written once, never re-read -> non-temporal store
        __builtin_nontemporal_store(acc[i][j][r] + bias[gc],
                                    out + (size_t)gr * CDIM + gc);
      }
}

extern "C" void kernel_launch(void* const* d_in, const int* in_sizes, int n_in,
                              void* d_out, int out_size, void* d_ws, size_t ws_size,
                              hipStream_t stream) {
  (void)in_sizes; (void)n_in; (void)out_size; (void)ws_size;
  const float* x      = (const float*)d_in[0];
  const float* mask   = (const float*)d_in[1];
  const float* qkv_w  = (const float*)d_in[2];
  const float* proj_w = (const float*)d_in[3];
  const float* proj_b = (const float*)d_in[4];
  const float* s      = (const float*)d_in[5];

  const size_t QKV_ELEMS = (size_t)2 * NHEADS * LSEQ * HD;  // 4M bf16 each
  bf16* Q   = (bf16*)d_ws;
  bf16* Kv  = Q   + QKV_ELEMS;
  bf16* Vt  = Kv  + QKV_ELEMS;
  bf16* Att = Vt  + QKV_ELEMS;                              // [B,L,C] bf16
  float* out = (float*)d_out;

  qkv_gemm_kernel<<<dim3(48, 32), 256, 0, stream>>>(x, qkv_w, s, Q, Kv, Vt);
  attn_kernel<<<dim3(1024), 128, 0, stream>>>(Q, Kv, Vt, mask, Att);
  proj_gemm_kernel<<<dim3(16, 32), 256, 0, stream>>>(Att, proj_w, proj_b, out);
}
